// MaskedAttention_69475390980786
// MI455X (gfx1250) — compile-verified, matrix operand order checked by reading, not run on hardware
//
#include <hip/hip_runtime.h>
#include <hip/hip_bf16.h>

// MaskedAttention on MI455X (gfx1250, wave32, WMMA f16 16x16x32, f32 accum)
// All WMMA fragments are loaded as contiguous b128 accesses via a bit-3<->4
// swizzled, fragment-major tile layout.  mask_attn (256 MB) streamed once.

typedef __attribute__((ext_vector_type(16))) _Float16 v16h;
typedef __attribute__((ext_vector_type(8)))  _Float16 v8h;
typedef __attribute__((ext_vector_type(4)))  _Float16 v4h;
typedef __attribute__((ext_vector_type(8)))  float    v8f;

#define B_  4
#define N_  1024
#define C_  1024
#define H_  16
#define HD_ 64
#define SCALE_ 0.125f   // 64^-0.5

// swap bits 3<->4 of a 5-bit index: makes a lane's 16 fragment halves
// (k = base..base+7, base+16..base+23) contiguous in the stored tile row.
__device__ __forceinline__ int swz(int k) {
  return (k & 7) | ((k & 8) << 1) | ((k & 16) >> 1);
}
__device__ __forceinline__ v16h cat(v8h lo, v8h hi) {
  return __builtin_shufflevector(lo, hi, 0,1,2,3,4,5,6,7,8,9,10,11,12,13,14,15);
}
__device__ __forceinline__ v8f wmma16(v16h a, v16h b, v8f c) {
  return __builtin_amdgcn_wmma_f32_16x16x32_f16(false, a, false, b, (short)0, c, false, false);
}

// ---------------------------------------------------------------- cvt kernel
__global__ __launch_bounds__(256) void cvt_f32_f16(const float* __restrict__ in,
                                                   _Float16* __restrict__ out, int n4) {
  int i = blockIdx.x * 256 + threadIdx.x;
  if (i < n4) {
    float4 v = ((const float4*)in)[i];
    v4h r; r[0] = (_Float16)v.x; r[1] = (_Float16)v.y;
           r[2] = (_Float16)v.z; r[3] = (_Float16)v.w;
    ((v4h*)out)[i] = r;
  }
}

// --------------------------------------------------------------- QKV GEMM
// out[m,c] = sum_k X[m,k]*W[c,k] + bias[c]; epilogue applies mask_param and
// q-scale, scatters q/k to [B,H,N,HD] and v TRANSPOSED to [B,H,HD,N].
__global__ __launch_bounds__(128) void qkv_gemm_kernel(
    const _Float16* __restrict__ X, const _Float16* __restrict__ W,
    const float* __restrict__ bias, const float* __restrict__ mp,
    _Float16* __restrict__ qo, _Float16* __restrict__ ko, _Float16* __restrict__ vo)
{
  __shared__ __align__(16) _Float16 bsw[64][56];   // 64 cols x 32 k (swizzled)
  const int tid = threadIdx.x, lane = tid & 31, wv = tid >> 5;
  const int ln = lane & 15, g8 = lane >> 4;
  const int row0 = blockIdx.y * 64, col0 = blockIdx.x * 64;

  const v8h* arow = (const v8h*)(X + (size_t)(row0 + wv*16 + ln) * C_);

  const v8f vz = {0.f,0.f,0.f,0.f,0.f,0.f,0.f,0.f};
  v8f acc[4] = {vz, vz, vz, vz};

  for (int c32 = 0; c32 < 32; ++c32) {           // k chunk = c32*32
    __syncthreads();
#pragma unroll
    for (int i = 0; i < 4; ++i) {                // fill B tile, b64 granules
      int gid = tid + i*128;
      int c = gid >> 3, k4 = (gid & 7) * 4;
      uint2 w4 = *(const uint2*)(W + (size_t)(col0 + c) * C_ + c32*32 + k4);
      *(uint2*)&bsw[c][swz(k4)] = w4;
    }
    __syncthreads();
    v16h a = cat(arow[c32*4 + g8], arow[c32*4 + 2 + g8]);   // 2x b128 global
#pragma unroll
    for (int t = 0; t < 4; ++t) {
      const v8h* bp = (const v8h*)&bsw[t*16 + ln][0];
      v16h b = cat(bp[g8*2], bp[g8*2 + 1]);                 // 2x ds b128
      acc[t] = wmma16(a, b, acc[t]);
    }
  }

#pragma unroll
  for (int t = 0; t < 4; ++t) {
    int col = col0 + t*16 + ln;
    int which = col >> 10, rem = col & 1023;
    int hh = rem >> 6, dd = rem & 63;
    float scale = mp[dd] * (which == 0 ? SCALE_ : 1.0f);
    float bcol = bias[col];
#pragma unroll
    for (int r = 0; r < 8; ++r) {
      int row = row0 + wv*16 + r + 8*g8;
      int bb = row >> 10, nn = row & 1023;
      float val = (acc[t][r] + bcol) * scale;
      if (which == 2) {  // V transposed: [bh][d][n]
        vo[((size_t)(bb*H_ + hh) * HD_ + dd) * N_ + nn] = (_Float16)val;
      } else {
        _Float16* dst = (which == 0) ? qo : ko;
        dst[((size_t)(bb*H_ + hh) * N_ + nn) * HD_ + dd] = (_Float16)val;
      }
    }
  }
}

// ------------------------------------------------- fused flash attention
__global__ __launch_bounds__(128) void flash_attn_kernel(
    const _Float16* __restrict__ Q, const _Float16* __restrict__ K,
    const _Float16* __restrict__ Vt, const float* __restrict__ mask_attn,
    const float* __restrict__ blend, _Float16* __restrict__ O)
{
  __shared__ __align__(16) _Float16 ktsw[64][72];     // key-major, d swizzled
  __shared__ __align__(16) _Float16 vtsw[64][72];     // d-major, key swizzled
  __shared__ __align__(16) _Float16 pbuf[4][16][72];  // qrow-major, k swizzled

  const int tid = threadIdx.x, lane = tid & 31, wv = tid >> 5;
  const int ln = lane & 15, g8 = lane >> 4;
  const int bh = blockIdx.y;            // b*H + h
  const int b_ = bh >> 4, h_ = bh & 15;
  const int q0 = blockIdx.x * 64 + wv * 16;
  const size_t bhN = (size_t)bh * N_;

  // Q A-fragments (pre-scaled by SCALE*mask_param), direct b128 global loads
  const v8h* qp = (const v8h*)(Q + (bhN + q0 + ln) * HD_);
  v16h qa0 = cat(qp[g8],     qp[2 + g8]);
  v16h qa1 = cat(qp[4 + g8], qp[6 + g8]);

  const v8f vz = {0.f,0.f,0.f,0.f,0.f,0.f,0.f,0.f};
  v8f o2[4] = {vz,vz,vz,vz}, o1[4] = {vz,vz,vz,vz};
  float mrow[8], lrow[8], den2[8];
#pragma unroll
  for (int r = 0; r < 8; ++r) { mrow[r] = -1e30f; lrow[r] = 0.f; den2[r] = 0.f; }
  float den1 = 0.f;                     // A-layout row (ln) partial

  const float* mbase = mask_attn + (bhN + q0 + ln) * N_;   // A-frag row
  const float* bbase = blend + (size_t)(q0 + ln) * N_;

  for (int kb = 0; kb < N_; kb += 64) {
    __syncthreads();
#pragma unroll
    for (int i = 0; i < 8; ++i) {       // fill K,V tiles, b64 granules
      int gid = tid + i*128;
      int row = gid >> 4, c4 = (gid & 15) * 4;
      int pos = (c4 & 32) + swz(c4 & 31);
      uint2 kk4 = *(const uint2*)(K  + (bhN + kb + row) * HD_ + c4);
      *(uint2*)&ktsw[row][pos] = kk4;
      uint2 vv4 = *(const uint2*)(Vt + ((size_t)bh * HD_ + row) * N_ + kb + c4);
      *(uint2*)&vtsw[row][pos] = vv4;
    }
    // prefetch next mask_attn block while we compute
    if (kb + 64 < N_) __builtin_prefetch(mbase + kb + 64, 0, 1);
    __syncthreads();

    // S = Q @ K^T  (16 x 64)
    v8f s[4];
#pragma unroll
    for (int t = 0; t < 4; ++t) {
      const v8h* kp = (const v8h*)&ktsw[t*16 + ln][0];
      v16h b0 = cat(kp[g8*2],     kp[g8*2 + 1]);
      v16h b1 = cat(kp[4 + g8*2], kp[4 + g8*2 + 1]);
      s[t] = vz;
      s[t] = wmma16(qa0, b0, s[t]);
      s[t] = wmma16(qa1, b1, s[t]);
    }

    // online softmax row max (rows live in 16-lane groups)
    float alpha[8], mnew[8];
#pragma unroll
    for (int r = 0; r < 8; ++r) {
      float mx = s[0][r];
#pragma unroll
      for (int t = 1; t < 4; ++t) mx = fmaxf(mx, s[t][r]);
      mx = fmaxf(mx, __shfl_xor(mx, 1, 32));
      mx = fmaxf(mx, __shfl_xor(mx, 2, 32));
      mx = fmaxf(mx, __shfl_xor(mx, 4, 32));
      mx = fmaxf(mx, __shfl_xor(mx, 8, 32));
      mnew[r]  = fmaxf(mrow[r], mx);
      alpha[r] = __expf(mrow[r] - mnew[r]);
      mrow[r]  = mnew[r];
    }

    // p = exp(s-m); p2 = p*(1-blend) -> swizzled pbuf (C->A transpose)
#pragma unroll
    for (int r = 0; r < 8; ++r) {
      int qn = q0 + r + 8*g8;
      float psum = 0.f, p2sum = 0.f;
#pragma unroll
      for (int t = 0; t < 4; ++t) {
        int k16 = t*16 + ln;
        float bl = blend[(size_t)qn * N_ + kb + k16];
        float p  = __expf(s[t][r] - mnew[r]);
        float p2 = p * (1.0f - bl);
        psum += p; p2sum += p2;
        pbuf[wv][r + 8*g8][(k16 & 32) + swz(k16 & 31)] = (_Float16)p2;
      }
      lrow[r] = lrow[r] * alpha[r] + psum;
      den2[r] = den2[r] * alpha[r] + p2sum;
#pragma unroll
      for (int t = 0; t < 4; ++t) o2[t][r] *= alpha[r];
    }

    // T2 += P2 @ V ; T1 += (blend*mask_attn) @ V
#pragma unroll
    for (int cb = 0; cb < 2; ++cb) {
      const v8h* pp = (const v8h*)&pbuf[wv][ln][0];
      v16h pa = cat(pp[cb*4 + g8*2], pp[cb*4 + g8*2 + 1]);

      // W1 A-fragment: 4x float4 each from mask_attn / blend (b128 loads)
      const float4* m4 = (const float4*)(mbase + kb + cb*32);
      const float4* b4 = (const float4*)(bbase + kb + cb*32);
      float4 ma0 = m4[g8*2], ma1 = m4[g8*2+1], ma2 = m4[4+g8*2], ma3 = m4[4+g8*2+1];
      float4 bl0 = b4[g8*2], bl1 = b4[g8*2+1], bl2 = b4[4+g8*2], bl3 = b4[4+g8*2+1];
      float wf[16] = {
        ma0.x*bl0.x, ma0.y*bl0.y, ma0.z*bl0.z, ma0.w*bl0.w,
        ma1.x*bl1.x, ma1.y*bl1.y, ma1.z*bl1.z, ma1.w*bl1.w,
        ma2.x*bl2.x, ma2.y*bl2.y, ma2.z*bl2.z, ma2.w*bl2.w,
        ma3.x*bl3.x, ma3.y*bl3.y, ma3.z*bl3.z, ma3.w*bl3.w };
      v16h wa;
#pragma unroll
      for (int e = 0; e < 16; ++e) { den1 += wf[e]; wa[e] = (_Float16)wf[e]; }

#pragma unroll
      for (int t = 0; t < 4; ++t) {
        const v8h* vp = (const v8h*)&vtsw[t*16 + ln][0];
        v16h vb = cat(vp[cb*4 + g8*2], vp[cb*4 + g8*2 + 1]);
        o2[t] = wmma16(pa, vb, o2[t]);
        o1[t] = wmma16(wa, vb, o1[t]);
      }
    }
  }

  // final cross-lane reductions
#pragma unroll
  for (int r = 0; r < 8; ++r) {
#pragma unroll
    for (int sft = 1; sft <= 8; sft <<= 1) {
      lrow[r] += __shfl_xor(lrow[r], sft, 32);
      den2[r] += __shfl_xor(den2[r], sft, 32);
    }
  }
  den1 += __shfl_xor(den1, 16, 32);
  float den1c[8];
#pragma unroll
  for (int r = 0; r < 8; ++r)
    den1c[r] = __shfl(den1, 8*g8 + r, 32);

  // out = (T1 + T2/l) / max(den1 + den2/l, eps) -> [B,N,H*HD] f16
#pragma unroll
  for (int t = 0; t < 4; ++t) {
    int d = t*16 + ln;
#pragma unroll
    for (int r = 0; r < 8; ++r) {
      int qn = q0 + r + 8*g8;
      float linv = 1.0f / lrow[r];
      float num  = o1[t][r] + o2[t][r] * linv;
      float den  = fmaxf(den1c[r] + den2[r] * linv, 1e-12f);
      O[((size_t)b_ * N_ + qn) * C_ + h_*HD_ + d] = (_Float16)(num / den);
    }
  }
}

// --------------------------------------------------------------- proj GEMM
__global__ __launch_bounds__(128) void proj_gemm_kernel(
    const _Float16* __restrict__ X, const _Float16* __restrict__ W,
    const float* __restrict__ bias, float* __restrict__ out)
{
  __shared__ __align__(16) _Float16 bsw[64][56];
  const int tid = threadIdx.x, lane = tid & 31, wv = tid >> 5;
  const int ln = lane & 15, g8 = lane >> 4;
  const int row0 = blockIdx.y * 64, col0 = blockIdx.x * 64;

  const v8h* arow = (const v8h*)(X + (size_t)(row0 + wv*16 + ln) * C_);

  const v8f vz = {0.f,0.f,0.f,0.f,0.f,0.f,0.f,0.f};
  v8f acc[4] = {vz, vz, vz, vz};

  for (int c32 = 0; c32 < 32; ++c32) {
    __syncthreads();
#pragma unroll
    for (int i = 0; i < 4; ++i) {
      int gid = tid + i*128;
      int c = gid >> 3, k4 = (gid & 7) * 4;
      uint2 w4 = *(const uint2*)(W + (size_t)(col0 + c) * C_ + c32*32 + k4);
      *(uint2*)&bsw[c][swz(k4)] = w4;
    }
    __syncthreads();
    v16h a = cat(arow[c32*4 + g8], arow[c32*4 + 2 + g8]);
#pragma unroll
    for (int t = 0; t < 4; ++t) {
      const v8h* bp = (const v8h*)&bsw[t*16 + ln][0];
      v16h b = cat(bp[g8*2], bp[g8*2 + 1]);
      acc[t] = wmma16(a, b, acc[t]);
    }
  }

#pragma unroll
  for (int t = 0; t < 4; ++t) {
    int col = col0 + t*16 + ln;
    float bcol = bias[col];
#pragma unroll
    for (int r = 0; r < 8; ++r) {
      int row = row0 + wv*16 + r + 8*g8;
      out[(size_t)row * C_ + col] = acc[t][r] + bcol;
    }
  }
}

// ---------------------------------------------------------------- launcher
extern "C" void kernel_launch(void* const* d_in, const int* in_sizes, int n_in,
                              void* d_out, int out_size, void* d_ws, size_t ws_size,
                              hipStream_t stream) {
  const float* mask_attn = (const float*)d_in[0];
  const float* blend     = (const float*)d_in[1];
  const float* x         = (const float*)d_in[2];
  const float* qkv_w     = (const float*)d_in[3];
  const float* qkv_b     = (const float*)d_in[4];
  const float* proj_w    = (const float*)d_in[5];
  const float* proj_b    = (const float*)d_in[6];
  const float* mask_p    = (const float*)d_in[7];
  float* out = (float*)d_out;

  char* ws = (char*)d_ws;                               // 48 MB used
  _Float16* xf   = (_Float16*)(ws);                     // 8 MB
  _Float16* wqkv = (_Float16*)(ws + ( 8u << 20));       // 6 MB
  _Float16* wpr  = (_Float16*)(ws + (14u << 20));       // 2 MB
  _Float16* qws  = (_Float16*)(ws + (16u << 20));       // 8 MB [B,H,N,HD]
  _Float16* kws  = (_Float16*)(ws + (24u << 20));       // 8 MB [B,H,N,HD]
  _Float16* vws  = (_Float16*)(ws + (32u << 20));       // 8 MB [B,H,HD,N] (transposed)
  _Float16* aout = (_Float16*)(ws + (40u << 20));       // 8 MB [B,N,C]

  int n4;
  n4 = (B_ * N_ * C_) / 4; cvt_f32_f16<<<(n4 + 255) / 256, 256, 0, stream>>>(x, xf, n4);
  n4 = (3 * C_ * C_) / 4;  cvt_f32_f16<<<(n4 + 255) / 256, 256, 0, stream>>>(qkv_w, wqkv, n4);
  n4 = (C_ * C_) / 4;      cvt_f32_f16<<<(n4 + 255) / 256, 256, 0, stream>>>(proj_w, wpr, n4);

  qkv_gemm_kernel  <<<dim3(48, 64), 128, 0, stream>>>(xf, wqkv, qkv_b, mask_p, qws, kws, vws);
  flash_attn_kernel<<<dim3(16, 64), 128, 0, stream>>>(qws, kws, vws, mask_attn, blend, aout);
  proj_gemm_kernel <<<dim3(16, 64), 128, 0, stream>>>(aout, wpr, proj_b, out);
}
